// SequencePartition_20950850470486
// MI455X (gfx1250) — matrix operand order repaired and checked by reading, not compiled, not run on hardware
//
#include <hip/hip_runtime.h>
#include <hip/hip_bf16.h>

typedef unsigned short u16;
typedef __attribute__((ext_vector_type(8)))  __bf16 v8bf;
typedef __attribute__((ext_vector_type(16))) __bf16 v16bf;
typedef __attribute__((ext_vector_type(8)))  float  v8f;
typedef __attribute__((ext_vector_type(4)))  int    v4i;
typedef __attribute__((address_space(1)))    v4i    as1_v4i;
typedef __attribute__((address_space(3)))    v4i    as3_v4i;

// dims
#define B_  32
#define S_  2048
#define E_  512
#define N_  256
#define P_  16
#define K_  16
#define KDIM 8192      // E_*K_
#define M_  8192       // B_*N_
#define LDA 40         // padded LDS row stride (u16): 80B rows, 16B-aligned segments
#define NCHUNK 256     // KDIM/32

#if __has_builtin(__builtin_amdgcn_global_load_async_to_lds_b128) && \
    __has_builtin(__builtin_amdgcn_s_wait_asynccnt)
#define HAVE_ASYNC 1
#define WAIT_ASYNC(n) __builtin_amdgcn_s_wait_asynccnt(n)
#else
#define HAVE_ASYNC 0
#define WAIT_ASYNC(n)
#endif

static __device__ __forceinline__ u16 f2bf(float f) {
    unsigned u = __float_as_uint(f);
    unsigned r = (u + 0x7FFFu + ((u >> 16) & 1u)) >> 16;   // RNE
    return (u16)r;
}

// ---- pack conv_w [E,E,K] f32 -> Wp[e_out][k*512+e_in] bf16 (8MB, L2-resident) ----
__global__ __launch_bounds__(256) void pack_w(const float* __restrict__ cw, u16* __restrict__ Wp) {
    int t = blockIdx.x * 256 + threadIdx.x;          // 0 .. 512*8192-1
    int eo  = t >> 13;
    int kd  = t & (KDIM - 1);
    int ein = kd & (E_ - 1);
    int k   = kd >> 9;
    Wp[t] = f2bf(cw[(size_t)eo * KDIM + ein * K_ + k]);
}

// ---- convert x to bf16 once (64MB, L2-resident) ----
__global__ __launch_bounds__(256) void conv_xbf(const float* __restrict__ x, u16* __restrict__ xbf) {
    size_t t = ((size_t)blockIdx.x * 256 + threadIdx.x) * 8;
    float4 a = *(const float4*)(x + t);
    float4 b = *(const float4*)(x + t + 4);
    uint4 o;
    o.x = (unsigned)f2bf(a.x) | ((unsigned)f2bf(a.y) << 16);
    o.y = (unsigned)f2bf(a.z) | ((unsigned)f2bf(a.w) << 16);
    o.z = (unsigned)f2bf(b.x) | ((unsigned)f2bf(b.y) << 16);
    o.w = (unsigned)f2bf(b.z) | ((unsigned)f2bf(b.w) << 16);
    *(uint4*)(xbf + t) = o;
}

// ---- seed off[m][c] = lin_b[c]; atomics accumulate on top ----
__global__ __launch_bounds__(256) void init_off(const float* __restrict__ lin_b, float* __restrict__ off) {
    int t = blockIdx.x * 256 + threadIdx.x;
    off[t] = lin_b[t & 1];
}

// stage one A tile (256 rows x 32 K, bf16) into LDS buffer.
// 1024 units of 16B; 4 units/thread; exactly 4 async issues per wave (uniform ASYNCcnt math).
template<bool XBF>
static __device__ __forceinline__ void stage_tile(
    u16* As, const float* x, const u16* xbf, int b, int chunk, int tid)
{
    const int k    = chunk >> 4;            // conv tap
    const int ein0 = (chunk & 15) * 32;     // e_in base
    #pragma unroll
    for (int i = 0; i < 4; i++) {
        int u   = tid + i * 256;
        int row = u >> 2;                   // n (block == full batch)
        int seg = u & 3;
        int s   = row * 8 - 4 + k;          // STRIDE=8, PAD=4
        bool valid = (s >= 0) && (s < S_);
        u16* dst = As + row * LDA + seg * 8;
        if constexpr (XBF) {
#if HAVE_ASYNC
            if (valid) {
                const u16* g = xbf + ((size_t)(b * S_ + s)) * E_ + ein0 + seg * 8;
                __builtin_amdgcn_global_load_async_to_lds_b128(
                    (as1_v4i*)g, (as3_v4i*)dst, 0, 0);
            } else {
                uint4 z = {0u, 0u, 0u, 0u};
                *(uint4*)dst = z;
            }
#else
            uint4 v = {0u, 0u, 0u, 0u};
            if (valid) v = *(const uint4*)(xbf + ((size_t)(b * S_ + s)) * E_ + ein0 + seg * 8);
            *(uint4*)dst = v;
#endif
        } else {
            uint4 pk = {0u, 0u, 0u, 0u};
            if (valid) {
                const float* gp = x + ((size_t)(b * S_ + s)) * E_ + ein0 + seg * 8;
                float4 f0 = *(const float4*)gp;
                float4 f1 = *(const float4*)(gp + 4);
                pk.x = (unsigned)f2bf(f0.x) | ((unsigned)f2bf(f0.y) << 16);
                pk.y = (unsigned)f2bf(f0.z) | ((unsigned)f2bf(f0.w) << 16);
                pk.z = (unsigned)f2bf(f1.x) | ((unsigned)f2bf(f1.y) << 16);
                pk.w = (unsigned)f2bf(f1.z) | ((unsigned)f2bf(f1.w) << 16);
            }
            *(uint4*)dst = pk;
        }
    }
}

// ---- conv-as-GEMM (bf16 WMMA, f32 accum), double-buffered async A staging,
//      fused relu->Linear(E,2) epilogue.  grid (4 Nblk, 32 batches), 256 thr = 8 waves (4M x 2N),
//      wave tile 64x64 -> 16 WMMA/chunk/wave. ----
template<bool XBF>
__global__ __launch_bounds__(256) void conv_gemm_off(
    const float* __restrict__ x, const u16* __restrict__ xbf,
    const u16* __restrict__ Wp,
    const float* __restrict__ conv_b, const float* __restrict__ lin_w,
    float* __restrict__ off)
{
    __shared__ __align__(16) u16 As[2][256 * LDA];   // 2 x 20KB

    const int tid  = threadIdx.x;
    const int lane = tid & 31;
    const int wid  = tid >> 5;
    const int waveM = (wid & 3) * 64;     // 4 wave rows
    const int waveN = (wid >> 2) * 64;    // 2 wave cols
    const int Nb = blockIdx.x;            // 0..3  (e_out block of 128)
    const int b  = blockIdx.y;            // 0..31 (one full batch: rows m = b*256 .. +255)

    const int hl   = lane >> 4;
    const int l16  = lane & 15;
    const int kloA = hl * 16;             // A frag: K 0-15 vs 16-31 contiguous
    const int kb0  = hl * 16;             // B frag: K 0-15 vs 16-31 contiguous

    v8f acc[4][4];
    #pragma unroll
    for (int i = 0; i < 4; i++)
        #pragma unroll
        for (int j = 0; j < 4; j++) acc[i][j] = (v8f)(0.0f);

    int colB[4];
    #pragma unroll
    for (int j = 0; j < 4; j++) colB[j] = Nb * 128 + waveN + j * 16 + l16;

    stage_tile<XBF>(As[0], x, xbf, b, 0, tid);

    for (int c = 0; c < NCHUNK; ++c) {
        if (c + 1 < NCHUNK) {
            stage_tile<XBF>(As[(c + 1) & 1], x, xbf, b, c + 1, tid);
            WAIT_ASYNC(4);                // previous fill (in-order) retired
        } else {
            WAIT_ASYNC(0);
        }
        __syncthreads();

        const u16* Ab = As[c & 1];

        // load all 8 fragments up-front (distinct regs -> clause'd ds/global loads,
        // then 16 back-to-back WMMAs without per-group s_wait_dscnt)
        v16bf bfrag[4];
        #pragma unroll
        for (int j = 0; j < 4; j++) {
            const u16* bp = Wp + (size_t)colB[j] * KDIM + c * 32 + kb0;
            v8bf lo = *(const v8bf*)bp;
            v8bf hi = *(const v8bf*)(bp + 8);
            bfrag[j] = __builtin_shufflevector(lo, hi, 0,1,2,3,4,5,6,7,8,9,10,11,12,13,14,15);
        }
        v16bf afrag[4];
        #pragma unroll
        for (int i = 0; i < 4; i++) {
            const u16* ap = Ab + (waveM + i * 16 + l16) * LDA + kloA;
            v8bf lo = *(const v8bf*)ap;
            v8bf hi = *(const v8bf*)(ap + 8);
            afrag[i] = __builtin_shufflevector(lo, hi, 0,1,2,3,4,5,6,7,8,9,10,11,12,13,14,15);
        }
        #pragma unroll
        for (int i = 0; i < 4; i++)
            #pragma unroll
            for (int j = 0; j < 4; j++)
                acc[i][j] = __builtin_amdgcn_wmma_f32_16x16x32_bf16(
                    false, afrag[i], false, bfrag[j], (short)0, acc[i][j], false, false);
        __syncthreads();
    }

    // fused epilogue: off[m][c] += sum_e relu(y+conv_b)*lin_w[c][e] over this wave's 64 cols
    float cb[4], w0[4], w1[4];
    #pragma unroll
    for (int j = 0; j < 4; j++) {
        int eo = colB[j];
        cb[j] = conv_b[eo];
        w0[j] = lin_w[eo];
        w1[j] = lin_w[E_ + eo];
    }
    #pragma unroll
    for (int i = 0; i < 4; i++) {
        #pragma unroll
        for (int r = 0; r < 8; r++) {
            float s0 = 0.f, s1 = 0.f;
            #pragma unroll
            for (int j = 0; j < 4; j++) {
                float v = fmaxf(acc[i][j][r] + cb[j], 0.0f);
                s0 += v * w0[j];
                s1 += v * w1[j];
            }
            #pragma unroll
            for (int d = 1; d < 16; d <<= 1) {       // reduce 16 cols per lane half
                s0 += __shfl_xor(s0, d, 32);
                s1 += __shfl_xor(s1, d, 32);
            }
            if (l16 == 0) {                          // lanes 0 (row r) / 16 (row r+8)
                int m = b * N_ + waveM + i * 16 + r + hl * 8;
                atomicAdd(&off[m * 2 + 0], s0);
                atomicAdd(&off[m * 2 + 1], s1);
            }
        }
    }
}

// ---- decode boxes + nearest gather: one block per (b,n), 16x512 f32 rows via b128 ----
__global__ __launch_bounds__(256) void decode_gather(
    const float* __restrict__ x, const float* __restrict__ off,
    const float* __restrict__ wbias, float* __restrict__ out)
{
    __shared__ int sidx[P_];
    const int m = blockIdx.x;             // 0..8191
    const int b = m >> 8, n = m & (N_ - 1);
    const int tid = threadIdx.x;

    if (tid < P_) {
        float o0 = off[m * 2 + 0], o1 = off[m * 2 + 1];
        float poi    = 1.0f / (float)N_;
        float anchor = (0.5f + (float)n) * poi;
        float dx = tanhf(o0) * poi;
        float dw = fmaxf(tanhf(o1 + wbias[0]), 0.0f) * poi;
        float x0 = fminf(fmaxf(anchor + dx - dw, 0.0f), 1.0f);
        float x1 = fminf(fmaxf(anchor + dx + dw, 0.0f), 1.0f);
        float t  = (float)tid * (1.0f / (float)(P_ - 1));
        float gx = x0 * (1.0f - t) + x1 * t;
        float ix = gx * (float)S_ - 0.5f;
        int id = (int)rintf(ix);                   // RNE matches jnp.round
        sidx[tid] = min(max(id, 0), S_ - 1);
    }
    __syncthreads();

    const float4* xb = (const float4*)(x + (size_t)b * S_ * E_);
    float4* ob = (float4*)(out + (size_t)m * P_ * E_);
    #pragma unroll
    for (int i = 0; i < 8; i++) {
        int u  = tid + i * 256;                    // 0..2047
        int p  = u >> 7;
        int e4 = u & 127;
        ob[u] = xb[(size_t)sidx[p] * 128 + e4];
    }
}

extern "C" void kernel_launch(void* const* d_in, const int* in_sizes, int n_in,
                              void* d_out, int out_size, void* d_ws, size_t ws_size,
                              hipStream_t stream) {
    const float* x      = (const float*)d_in[0];
    const float* conv_w = (const float*)d_in[1];
    const float* conv_b = (const float*)d_in[2];
    const float* lin_w  = (const float*)d_in[3];
    const float* lin_b  = (const float*)d_in[4];
    const float* wbias  = (const float*)d_in[5];
    float* out = (float*)d_out;

    const size_t wpBytes = (size_t)E_ * KDIM * sizeof(u16);          // 8 MB
    const size_t offOff  = wpBytes;
    const size_t xbfOff  = offOff + (size_t)M_ * 2 * sizeof(float);  // +64 KB
    const size_t need    = xbfOff + (size_t)B_ * S_ * E_ * sizeof(u16);

    u16*   Wp  = (u16*)d_ws;
    float* off = (float*)((char*)d_ws + offOff);

    pack_w<<<(E_ * KDIM) / 256, 256, 0, stream>>>(conv_w, Wp);
    init_off<<<(M_ * 2) / 256, 256, 0, stream>>>(lin_b, off);

    if (ws_size >= need) {
        u16* xbf = (u16*)((char*)d_ws + xbfOff);
        conv_xbf<<<(B_ * S_ * E_) / (8 * 256), 256, 0, stream>>>(x, xbf);
        conv_gemm_off<true><<<dim3(4, 32), 256, 0, stream>>>(x, xbf, Wp, conv_b, lin_w, off);
    } else {
        conv_gemm_off<false><<<dim3(4, 32), 256, 0, stream>>>(x, nullptr, Wp, conv_b, lin_w, off);
    }
    decode_gather<<<M_, 256, 0, stream>>>(x, off, wbias, out);
}
